// MoE_67242007986667
// MI455X (gfx1250) — matrix operand order repaired and checked by reading, not compiled, not run on hardware
//
#include <hip/hip_runtime.h>
#include <hip/hip_bf16.h>
#include <math.h>

typedef __bf16 bf16;
typedef __attribute__((ext_vector_type(16))) __bf16 v16bf;
typedef __attribute__((ext_vector_type(8)))  __bf16 v8bf;
typedef __attribute__((ext_vector_type(4)))  __bf16 v4bf;
typedef __attribute__((ext_vector_type(8)))  float  v8f;

typedef __attribute__((address_space(3))) bf16 lds_bf16;

#define NTOK 2048
#define DIM  2048
#define NEXP 8
#define FSH  1536
#define FEX  768
#define NROUTED (NTOK * 2)   // exactly N * TOP_K gathered rows

// ---------------------------------------------------------------- helpers ---

__device__ __forceinline__ v8f wmma_bf16(v16bf a, v16bf b, v8f c) {
  // (neg_a, A, neg_b, B, c_mod, C, reuse_a, reuse_b)
  return __builtin_amdgcn_wmma_f32_16x16x32_bf16(false, a, false, b, (short)0, c,
                                                 false, false);
}

__device__ __forceinline__ v16bf cat8(v8bf lo, v8bf hi) {
  return __builtin_shufflevector(lo, hi, 0, 1, 2, 3, 4, 5, 6, 7,
                                 8, 9, 10, 11, 12, 13, 14, 15);
}

__device__ __forceinline__ float silu(float g) {
  return g * (1.0f / (1.0f + __expf(-g)));
}

// 16-byte async copy global -> LDS (GLOBAL_LOAD_ASYNC_TO_LDS_B128, ASYNCcnt).
__device__ __forceinline__ void async_copy16(bf16* ldsDst, const bf16* gSrc) {
  uint32_t loff = (uint32_t)(uintptr_t)(lds_bf16*)ldsDst;  // LDS byte offset
  asm volatile("global_load_async_to_lds_b128 %0, %1, off"
               :
               : "v"(loff), "v"(gSrc)
               : "memory");
}

__device__ __forceinline__ void wait_async0() {
  __builtin_amdgcn_s_wait_asynccnt(0);
}

// ------------------------------------------------------------ small kernels ---

__global__ __launch_bounds__(256) void convert_x_kernel(
    const float* __restrict__ x, bf16* __restrict__ xb) {
  int i = (blockIdx.x * 256 + threadIdx.x) * 4;
  float4 v = *(const float4*)(x + i);
  v4bf o;
  o[0] = (bf16)v.x; o[1] = (bf16)v.y; o[2] = (bf16)v.z; o[3] = (bf16)v.w;
  *(v4bf*)(xb + i) = o;
}

__global__ void init_kernel(int* __restrict__ counts) {
  if (threadIdx.x < NEXP) counts[threadIdx.x] = 0;
}

__global__ __launch_bounds__(256) void router_kernel(
    const float* __restrict__ x, const float* __restrict__ R,
    int* __restrict__ topk_id, float* __restrict__ topk_w,
    int* __restrict__ counts) {
  const int tok = blockIdx.x;
  const int tid = threadIdx.x;
  float acc[NEXP];
#pragma unroll
  for (int e = 0; e < NEXP; ++e) acc[e] = 0.0f;

  const float* xr = x + (size_t)tok * DIM;
  for (int d = tid; d < DIM; d += 256) {
    float xv = xr[d];
    const float* rr = R + (size_t)d * NEXP;
#pragma unroll
    for (int e = 0; e < NEXP; ++e) acc[e] += xv * rr[e];
  }

  __shared__ float red[256];
  __shared__ float logits[NEXP];
  for (int e = 0; e < NEXP; ++e) {
    red[tid] = acc[e];
    __syncthreads();
    for (int s = 128; s > 0; s >>= 1) {
      if (tid < s) red[tid] += red[tid + s];
      __syncthreads();
    }
    if (tid == 0) logits[e] = red[0];
    __syncthreads();
  }

  if (tid == 0) {
    float mx = logits[0];
#pragma unroll
    for (int e = 1; e < NEXP; ++e) mx = fmaxf(mx, logits[e]);
    float p[NEXP], sum = 0.0f;
#pragma unroll
    for (int e = 0; e < NEXP; ++e) { p[e] = __expf(logits[e] - mx); sum += p[e]; }
    float inv = 1.0f / sum;
#pragma unroll
    for (int e = 0; e < NEXP; ++e) p[e] *= inv;

    int e1 = 0;
#pragma unroll
    for (int e = 1; e < NEXP; ++e) if (p[e] > p[e1]) e1 = e;
    int e2 = (e1 == 0) ? 1 : 0;
#pragma unroll
    for (int e = 0; e < NEXP; ++e)
      if (e != e1 && p[e] > p[e2]) e2 = e;

    topk_id[tok * 2 + 0] = e1; topk_w[tok * 2 + 0] = p[e1];
    topk_id[tok * 2 + 1] = e2; topk_w[tok * 2 + 1] = p[e2];
    atomicAdd(&counts[e1], 1);
    atomicAdd(&counts[e2], 1);
  }
}

__global__ void scan_kernel(const int* __restrict__ counts,
                            int* __restrict__ offs, int* __restrict__ cursors) {
  if (threadIdx.x == 0) {
    int s = 0;
    for (int e = 0; e < NEXP; ++e) { offs[e] = s; s += counts[e]; }
  }
  if (threadIdx.x < NEXP) cursors[threadIdx.x] = 0;
}

__global__ __launch_bounds__(256) void assign_kernel(
    const int* __restrict__ topk_id, const float* __restrict__ topk_w,
    const int* __restrict__ offs, int* __restrict__ cursors,
    int* __restrict__ rowmap, float* __restrict__ roww) {
  int tok = blockIdx.x * 256 + threadIdx.x;
#pragma unroll
  for (int k = 0; k < 2; ++k) {
    int e = topk_id[tok * 2 + k];
    if (e < 0 || e >= NEXP) e = 0;
    int pos = offs[e] + atomicAdd(&cursors[e], 1);
    if (pos >= 0 && pos < NROUTED) {
      rowmap[pos] = tok;
      roww[pos] = topk_w[tok * 2 + k];
    }
  }
}

// ------------------------------------------------- fused gate+up WMMA GEMM ---
// C_g = X @ Wg, C_u = X @ Wu ; h = silu(C_g)*C_u  (bf16 out)
// Block tile 128x128, BK=32, 256 threads = 8 waves, wave = 32x64 region.

template <bool ROUTED>
__global__ __launch_bounds__(256) void gateup_kernel(
    const bf16* __restrict__ xb,       // [NTOK, DIM] bf16
    const float* __restrict__ Wg_all,  // [(E,) DIM, Fd] f32
    const float* __restrict__ Wu_all,
    bf16* __restrict__ hout,           // [rows, Fd] bf16
    const int* __restrict__ rowmap,
    const int* __restrict__ counts,
    const int* __restrict__ offsets,
    int Fd) {
  const int e = ROUTED ? blockIdx.z : 0;
  const int cnt = ROUTED ? counts[e] : NTOK;
  const int rowoff = ROUTED ? offsets[e] : 0;
  const int mblk = blockIdx.y * 128;
  if (ROUTED && mblk >= cnt) return;  // uniform early-exit
  const int fblk = blockIdx.x * 128;

  const float* Wg = Wg_all + (size_t)e * DIM * Fd;
  const float* Wu = Wu_all + (size_t)e * DIM * Fd;

  __shared__ __align__(16) bf16 ldsA[128][40];
  __shared__ __align__(16) bf16 ldsBg[128][40];  // [n][k] (transposed)
  __shared__ __align__(16) bf16 ldsBu[128][40];

  const int tid = threadIdx.x;
  const int lane = tid & 31;
  const int wave = tid >> 5;
  const int waveRow = wave & 3;   // 4 row groups * 32 rows
  const int waveCol = wave >> 2;  // 2 col groups * 64 cols
  const int half = lane >> 4;
  const int l16 = lane & 15;

  // A staging: thread -> (row, 32B half of a 64B row)
  const int ar = tid >> 1;
  const int ah = tid & 1;
  int tok;
  if (ROUTED) {
    int r = mblk + ar;
    int t = (r < cnt) ? rowmap[rowoff + r] : 0;
    if (t < 0 || t >= NTOK) t = 0;
    tok = t;
  } else {
    tok = mblk + ar;
  }
  const bf16* aptr = xb + (size_t)tok * DIM + ah * 16;
  bf16* aldst0 = &ldsA[ar][ah * 16 + 0];
  bf16* aldst1 = &ldsA[ar][ah * 16 + 8];

  // B staging: thread -> (k row, group of 16 f columns)
  const int bd = tid >> 3;
  const int bfc = (tid & 7) * 16;
  const float* gbase = Wg + (size_t)bd * Fd + fblk + bfc;
  const float* ubase = Wu + (size_t)bd * Fd + fblk + bfc;

  v8f accG[2][4], accU[2][4];
#pragma unroll
  for (int mi = 0; mi < 2; ++mi)
#pragma unroll
    for (int ni = 0; ni < 4; ++ni) { accG[mi][ni] = (v8f)0.0f; accU[mi][ni] = (v8f)0.0f; }

  for (int kb = 0; kb < DIM; kb += 32) {
    // stage A: async global->LDS copy (bf16 row-major, verbatim bytes)
    async_copy16(aldst0, aptr + kb + 0);
    async_copy16(aldst1, aptr + kb + 8);
    // stage B with fp32->bf16 conversion + transpose (VALU path)
    const float* gp = gbase + (size_t)kb * Fd;
    const float* up = ubase + (size_t)kb * Fd;
#pragma unroll
    for (int i = 0; i < 16; i += 4) {
      float4 g4 = *(const float4*)(gp + i);
      float4 u4 = *(const float4*)(up + i);
      ldsBg[bfc + i + 0][bd] = (bf16)g4.x; ldsBg[bfc + i + 1][bd] = (bf16)g4.y;
      ldsBg[bfc + i + 2][bd] = (bf16)g4.z; ldsBg[bfc + i + 3][bd] = (bf16)g4.w;
      ldsBu[bfc + i + 0][bd] = (bf16)u4.x; ldsBu[bfc + i + 1][bd] = (bf16)u4.y;
      ldsBu[bfc + i + 2][bd] = (bf16)u4.z; ldsBu[bfc + i + 3][bd] = (bf16)u4.w;
    }
    if (kb + 32 < DIM) {  // global_prefetch_b8 next weight tile
      __builtin_prefetch(gp + (size_t)32 * Fd, 0, 1);
      __builtin_prefetch(up + (size_t)32 * Fd, 0, 1);
    }
    wait_async0();   // this wave's async LDS writes are done
    __syncthreads(); // all waves' writes visible

    v16bf afr[2];
#pragma unroll
    for (int mi = 0; mi < 2; ++mi) {
      int r = waveRow * 32 + mi * 16 + l16;
      v8bf lo = *(const v8bf*)&ldsA[r][half * 16];
      v8bf hi = *(const v8bf*)&ldsA[r][half * 16 + 8];
      afr[mi] = cat8(lo, hi);
    }
#pragma unroll
    for (int ni = 0; ni < 4; ++ni) {
      int c = waveCol * 64 + ni * 16 + l16;
      v16bf bg = cat8(*(const v8bf*)&ldsBg[c][half * 16],
                      *(const v8bf*)&ldsBg[c][half * 16 + 8]);
      v16bf bu = cat8(*(const v8bf*)&ldsBu[c][half * 16],
                      *(const v8bf*)&ldsBu[c][half * 16 + 8]);
#pragma unroll
      for (int mi = 0; mi < 2; ++mi) {
        accG[mi][ni] = wmma_bf16(afr[mi], bg, accG[mi][ni]);
        accU[mi][ni] = wmma_bf16(afr[mi], bu, accU[mi][ni]);
      }
    }
    __syncthreads();
  }

  // epilogue: h = silu(G)*U, bf16 stores. C layout: row = j + half*8, col = l16.
#pragma unroll
  for (int mi = 0; mi < 2; ++mi) {
#pragma unroll
    for (int j = 0; j < 8; ++j) {
      int r = waveRow * 32 + mi * 16 + half * 8 + j;
      int rg = mblk + r;
      if (ROUTED && rg >= cnt) continue;
      size_t orow = (size_t)(ROUTED ? (rowoff + rg) : rg) * Fd + fblk;
#pragma unroll
      for (int ni = 0; ni < 4; ++ni) {
        float g = accG[mi][ni][j];
        float u = accU[mi][ni][j];
        hout[orow + waveCol * 64 + ni * 16 + l16] = (bf16)(silu(g) * u);
      }
    }
  }
}

// ------------------------------------------------------- down-proj WMMA GEMM ---
// y = h @ Wd ; shared: out = y (plain store, full coverage)
//             routed: out[tok] += w * y (atomic scatter)

template <bool ROUTED>
__global__ __launch_bounds__(256) void down_kernel(
    const bf16* __restrict__ h,        // [rows, Kd] bf16
    const float* __restrict__ Wd_all,  // [(E,) Kd, DIM] f32
    float* __restrict__ out,           // [NTOK, DIM] f32
    const int* __restrict__ rowmap,
    const float* __restrict__ roww,
    const int* __restrict__ counts,
    const int* __restrict__ offsets,
    int Kd) {
  const int e = ROUTED ? blockIdx.z : 0;
  const int cnt = ROUTED ? counts[e] : NTOK;
  const int rowoff = ROUTED ? offsets[e] : 0;
  const int mblk = blockIdx.y * 128;
  if (ROUTED && mblk >= cnt) return;
  const int cblk = blockIdx.x * 128;

  const float* Wd = Wd_all + (size_t)e * Kd * DIM;

  __shared__ __align__(16) bf16 ldsA[128][40];
  __shared__ __align__(16) bf16 ldsB[128][40];  // [n][k]

  const int tid = threadIdx.x;
  const int lane = tid & 31;
  const int wave = tid >> 5;
  const int waveRow = wave & 3;
  const int waveCol = wave >> 2;
  const int half = lane >> 4;
  const int l16 = lane & 15;

  const int ar = tid >> 1;
  const int ah = tid & 1;
  int hrow = mblk + ar;
  if (hrow >= cnt) hrow = 0;  // safe row; results masked in epilogue
  const bf16* aptr = h + (size_t)(rowoff + hrow) * Kd + ah * 16;
  bf16* aldst0 = &ldsA[ar][ah * 16 + 0];
  bf16* aldst1 = &ldsA[ar][ah * 16 + 8];

  const int bd = tid >> 3;
  const int bfc = (tid & 7) * 16;
  const float* bbase = Wd + (size_t)bd * DIM + cblk + bfc;

  v8f acc[2][4];
#pragma unroll
  for (int mi = 0; mi < 2; ++mi)
#pragma unroll
    for (int ni = 0; ni < 4; ++ni) acc[mi][ni] = (v8f)0.0f;

  for (int kb = 0; kb < Kd; kb += 32) {
    async_copy16(aldst0, aptr + kb + 0);
    async_copy16(aldst1, aptr + kb + 8);
    const float* bp = bbase + (size_t)kb * DIM;
#pragma unroll
    for (int i = 0; i < 16; i += 4) {
      float4 b4 = *(const float4*)(bp + i);
      ldsB[bfc + i + 0][bd] = (bf16)b4.x; ldsB[bfc + i + 1][bd] = (bf16)b4.y;
      ldsB[bfc + i + 2][bd] = (bf16)b4.z; ldsB[bfc + i + 3][bd] = (bf16)b4.w;
    }
    if (kb + 32 < Kd) __builtin_prefetch(bp + (size_t)32 * DIM, 0, 1);
    wait_async0();
    __syncthreads();

    v16bf afr[2];
#pragma unroll
    for (int mi = 0; mi < 2; ++mi) {
      int r = waveRow * 32 + mi * 16 + l16;
      v8bf lo = *(const v8bf*)&ldsA[r][half * 16];
      v8bf hi = *(const v8bf*)&ldsA[r][half * 16 + 8];
      afr[mi] = cat8(lo, hi);
    }
#pragma unroll
    for (int ni = 0; ni < 4; ++ni) {
      int c = waveCol * 64 + ni * 16 + l16;
      v16bf bfr = cat8(*(const v8bf*)&ldsB[c][half * 16],
                       *(const v8bf*)&ldsB[c][half * 16 + 8]);
#pragma unroll
      for (int mi = 0; mi < 2; ++mi)
        acc[mi][ni] = wmma_bf16(afr[mi], bfr, acc[mi][ni]);
    }
    __syncthreads();
  }

#pragma unroll
  for (int mi = 0; mi < 2; ++mi) {
#pragma unroll
    for (int j = 0; j < 8; ++j) {
      int r = waveRow * 32 + mi * 16 + half * 8 + j;
      int rg = mblk + r;
      if (ROUTED) {
        if (rg >= cnt) continue;
        int g = rowoff + rg;
        int tok = rowmap[g];
        if (tok < 0 || tok >= NTOK) tok = 0;
        float w = roww[g];
        size_t orow = (size_t)tok * DIM + cblk;
#pragma unroll
        for (int ni = 0; ni < 4; ++ni)
          atomicAdd(&out[orow + waveCol * 64 + ni * 16 + l16],
                    w * acc[mi][ni][j]);
      } else {
        size_t orow = (size_t)rg * DIM + cblk;
#pragma unroll
        for (int ni = 0; ni < 4; ++ni)
          out[orow + waveCol * 64 + ni * 16 + l16] = acc[mi][ni][j];
      }
    }
  }
}

// ------------------------------------------------------------------- launch ---

extern "C" void kernel_launch(void* const* d_in, const int* in_sizes, int n_in,
                              void* d_out, int out_size, void* d_ws,
                              size_t ws_size, hipStream_t stream) {
  const float* x  = (const float*)d_in[0];   // [2,1024,2048]
  const float* R  = (const float*)d_in[1];   // [2048,8]
  const float* sg = (const float*)d_in[2];   // [2048,1536]
  const float* su = (const float*)d_in[3];   // [2048,1536]
  const float* sd = (const float*)d_in[4];   // [1536,2048]
  const float* wg = (const float*)d_in[5];   // [8,2048,768]
  const float* wu = (const float*)d_in[6];   // [8,2048,768]
  const float* wd = (const float*)d_in[7];   // [8,768,2048]
  float* out = (float*)d_out;

  char* ws = (char*)d_ws;
  size_t off = 0;
  auto take = [&](size_t bytes) {
    char* p = ws + off;
    off = (off + bytes + 255) & ~(size_t)255;
    return p;
  };
  bf16*  xb      = (bf16*)take((size_t)NTOK * DIM * sizeof(bf16));
  bf16*  hsh     = (bf16*)take((size_t)NTOK * FSH * sizeof(bf16));
  bf16*  hrt     = (bf16*)take((size_t)NROUTED * FEX * sizeof(bf16));
  int*   tkid    = (int*)take((size_t)NTOK * 2 * sizeof(int));
  float* tkw     = (float*)take((size_t)NTOK * 2 * sizeof(float));
  int*   rowmap  = (int*)take((size_t)NROUTED * sizeof(int));
  float* roww    = (float*)take((size_t)NROUTED * sizeof(float));
  int*   counts  = (int*)take(NEXP * sizeof(int));
  int*   cursors = (int*)take(NEXP * sizeof(int));
  int*   offs    = (int*)take(NEXP * sizeof(int));
  (void)in_sizes; (void)n_in; (void)out_size; (void)ws_size;

  convert_x_kernel<<<(NTOK * DIM) / 1024, 256, 0, stream>>>(x, xb);
  init_kernel<<<1, 32, 0, stream>>>(counts);
  router_kernel<<<NTOK, 256, 0, stream>>>(x, R, tkid, tkw, counts);
  scan_kernel<<<1, 32, 0, stream>>>(counts, offs, cursors);
  assign_kernel<<<NTOK / 256, 256, 0, stream>>>(tkid, tkw, offs, cursors,
                                                rowmap, roww);

  // shared expert: h = silu(x@sg)*(x@su); out = h@sd (full coverage store)
  gateup_kernel<false><<<dim3(FSH / 128, NTOK / 128, 1), 256, 0, stream>>>(
      xb, sg, su, hsh, rowmap, counts, offs, FSH);
  down_kernel<false><<<dim3(DIM / 128, NTOK / 128, 1), 256, 0, stream>>>(
      hsh, sd, out, rowmap, roww, counts, offs, FSH);

  // routed experts: gathered rows per expert, scatter-add with top-k weights
  gateup_kernel<true><<<dim3(FEX / 128, NTOK / 128, NEXP), 256, 0, stream>>>(
      xb, wg, wu, hrt, rowmap, counts, offs, FEX);
  down_kernel<true><<<dim3(DIM / 128, NTOK / 128, NEXP), 256, 0, stream>>>(
      hrt, wd, out, rowmap, roww, counts, offs, FEX);
}
